// MixtralDecoderLayer_67293547594309
// MI455X (gfx1250) — matrix operand order repaired and checked by reading, not compiled, not run on hardware
//
#include <hip/hip_runtime.h>
#include <cstdint>

#define B_    2
#define S_    1024
#define H_    2048
#define NH_   16
#define NKV_  4
#define HD_   128
#define F_    4096
#define E_    8
#define T_    (B_ * S_)
#define QKVN_ 3072          // (NH + 2*NKV) * HD
#define EPS_  1e-5f
#define THETA_ 10000.0f

typedef __attribute__((ext_vector_type(16))) _Float16 v16h;
typedef __attribute__((ext_vector_type(8)))  float    v8f;
typedef int v4i_ __attribute__((vector_size(16)));

union FragA {
  v16h v;
  _Float16 h[16];
  uint32_t u[8];
};

__device__ inline v8f wmma_f16(v16h a, v16h b, v8f c) {
  // D = A(16x32 f16) * B(32x16 f16) + C(16x16 f32)
  return __builtin_amdgcn_wmma_f32_16x16x32_f16(
      /*neg_a=*/false, a, /*neg_b=*/false, b,
      /*c_mod=*/(short)0, c, /*reuse_a=*/false, /*reuse_b=*/false);
}

// Wave-local LDS producer->consumer fence (single-wave workgroups get S_NOP
// barriers, so use the CDNA5 split DS counter explicitly).
__device__ inline void wave_lds_fence() {
  __builtin_amdgcn_wave_barrier();
  asm volatile("s_wait_dscnt 0" ::: "memory");
  __builtin_amdgcn_wave_barrier();
}

// ---- async global->LDS (gfx1250 ASYNCcnt path) ----------------------------
#define AS1_ __attribute__((address_space(1)))
#define AS3_ __attribute__((address_space(3)))

#if defined(__has_builtin)
#if __has_builtin(__builtin_amdgcn_global_load_async_to_lds_b128) && \
    __has_builtin(__builtin_amdgcn_s_wait_asynccnt)
#define HAVE_ASYNC_LDS 1
#endif
#endif

__device__ inline void async_copy_b128(const _Float16* gsrc, _Float16* ldst) {
#ifdef HAVE_ASYNC_LDS
  __builtin_amdgcn_global_load_async_to_lds_b128(
      (AS1_ v4i_*)(AS1_ void*)gsrc, (AS3_ v4i_*)(AS3_ void*)ldst,
      /*offset=*/0, /*cpol=*/0);
#else
  *reinterpret_cast<uint4*>(ldst) = *reinterpret_cast<const uint4*>(gsrc);
#endif
}

__device__ inline void async_wait0() {
#ifdef HAVE_ASYNC_LDS
  __builtin_amdgcn_s_wait_asynccnt(0);
#endif
}

// ---- fragment loaders (ISA 7.12.2 layouts) --------------------------------
// A 16x32 f16: lanes 0-15 row M=lane, K pairs {0,1..6,7} in u[0..3], {16..23}
// in u[4..7]; lanes 16-31 same rows, K base +8.
__device__ inline v16h load_a_frag_lds(const _Float16* base, int ld) {
  const int lane = threadIdx.x & 31;
  const int m  = lane & 15;
  const int kb = (lane >> 4) * 8;
  const _Float16* rp = base + m * ld;
  FragA f;
#pragma unroll
  for (int j = 0; j < 8; ++j) {
    int k = ((j >> 2) << 4) + kb + ((j & 3) << 1);
    f.u[j] = *reinterpret_cast<const uint32_t*>(rp + k);
  }
  return f.v;
}

// B 32x16 f16 staged N-major in LDS: lane = column N, K pairs {0..7},{16..23};
// lanes 16-31 K base +8.
__device__ inline v16h load_b_frag_lds(const _Float16* base, int ld) {
  const int lane = threadIdx.x & 31;
  const int n  = lane & 15;
  const int kb = (lane >> 4) * 8;
  const _Float16* rp = base + n * ld + kb;
  FragA f;
#pragma unroll
  for (int j = 0; j < 8; ++j)
    f.u[j] = *reinterpret_cast<const uint32_t*>(rp + ((j >> 2) << 4) + ((j & 3) << 1));
  return f.v;
}

// ---- block GEMM core ------------------------------------------------------
// 256 threads = 8 waves. Block tile 128(M) x 64(N), BK=32.
// Wave grid 4x2, each wave owns a 32x32 macro-tile => 4 WMMAs / K-step.
// Double-buffered LDS, one barrier per K-step; A staged async to LDS.
#define ABUF_ (128 * 40)
#define BBUF_ (64 * 40)

__device__ inline void stage_a(const _Float16* __restrict__ A, int lda,
                               int rowBase, int k0, _Float16* As) {
  const int tid = threadIdx.x;
#pragma unroll
  for (int i = 0; i < 2; ++i) {
    int idx = tid + i * 256;           // 0..511 -> 128 rows x 4 chunks
    int row = idx >> 2;
    int cb  = (idx & 3) << 3;
    async_copy_b128(A + (size_t)(rowBase + row) * lda + k0 + cb,
                    As + row * 40 + cb);
  }
}

__device__ inline void stage_b(const float* __restrict__ Bm, int ldb,
                               int colBase, int k0, _Float16* Bs) {
  const int tid = threadIdx.x;
  int k  = tid >> 3;                   // 0..31
  int nb = (tid & 7) << 3;             // 0..56 step 8
  const float* src = Bm + (size_t)(k0 + k) * ldb + colBase + nb;
  float4 b0 = *reinterpret_cast<const float4*>(src);
  float4 b1 = *reinterpret_cast<const float4*>(src + 4);
  Bs[(nb + 0) * 40 + k] = (_Float16)b0.x;
  Bs[(nb + 1) * 40 + k] = (_Float16)b0.y;
  Bs[(nb + 2) * 40 + k] = (_Float16)b0.z;
  Bs[(nb + 3) * 40 + k] = (_Float16)b0.w;
  Bs[(nb + 4) * 40 + k] = (_Float16)b1.x;
  Bs[(nb + 5) * 40 + k] = (_Float16)b1.y;
  Bs[(nb + 6) * 40 + k] = (_Float16)b1.z;
  Bs[(nb + 7) * 40 + k] = (_Float16)b1.w;
}

__device__ inline void gemm_core(const _Float16* __restrict__ A, int lda,
                                 const float* __restrict__ Bm, int ldb,
                                 int K, int rowBase, int colBase,
                                 _Float16* As, _Float16* Bs,
                                 v8f (&acc)[2][2]) {
  const int w  = threadIdx.x >> 5;
  const int wm = w >> 1, wn = w & 1;

  stage_a(A, lda, rowBase, 0, As);
  stage_b(Bm, ldb, colBase, 0, Bs);
  async_wait0();
  __syncthreads();

  int p = 0;
  for (int k0 = 0; k0 < K; k0 += 32) {
    if (k0 + 32 < K) {
      stage_a(A, lda, rowBase, k0 + 32, As + (p ^ 1) * ABUF_);
      stage_b(Bm, ldb, colBase, k0 + 32, Bs + (p ^ 1) * BBUF_);
    }
    const _Float16* Ab = As + p * ABUF_;
    const _Float16* Bb = Bs + p * BBUF_;
    v16h a0 = load_a_frag_lds(Ab + (wm * 32) * 40, 40);
    v16h a1 = load_a_frag_lds(Ab + (wm * 32 + 16) * 40, 40);
    v16h b0 = load_b_frag_lds(Bb + (wn * 32) * 40, 40);
    v16h b1 = load_b_frag_lds(Bb + (wn * 32 + 16) * 40, 40);
    acc[0][0] = wmma_f16(a0, b0, acc[0][0]);
    acc[0][1] = wmma_f16(a0, b1, acc[0][1]);
    acc[1][0] = wmma_f16(a1, b0, acc[1][0]);
    acc[1][1] = wmma_f16(a1, b1, acc[1][1]);
    async_wait0();
    __syncthreads();
    p ^= 1;
  }
}

// dual-B version (fused w1/w3): 8 WMMAs / K-step
__device__ inline void gemm_core2(const _Float16* __restrict__ A, int lda,
                                  const float* __restrict__ B1m,
                                  const float* __restrict__ B3m, int ldb,
                                  int K, int rowBase, int colBase,
                                  _Float16* As, _Float16* Bs1, _Float16* Bs3,
                                  v8f (&acc1)[2][2], v8f (&acc3)[2][2]) {
  const int w  = threadIdx.x >> 5;
  const int wm = w >> 1, wn = w & 1;

  stage_a(A, lda, rowBase, 0, As);
  stage_b(B1m, ldb, colBase, 0, Bs1);
  stage_b(B3m, ldb, colBase, 0, Bs3);
  async_wait0();
  __syncthreads();

  int p = 0;
  for (int k0 = 0; k0 < K; k0 += 32) {
    if (k0 + 32 < K) {
      stage_a(A, lda, rowBase, k0 + 32, As + (p ^ 1) * ABUF_);
      stage_b(B1m, ldb, colBase, k0 + 32, Bs1 + (p ^ 1) * BBUF_);
      stage_b(B3m, ldb, colBase, k0 + 32, Bs3 + (p ^ 1) * BBUF_);
    }
    const _Float16* Ab  = As  + p * ABUF_;
    const _Float16* B1b = Bs1 + p * BBUF_;
    const _Float16* B3b = Bs3 + p * BBUF_;
    v16h a0  = load_a_frag_lds(Ab + (wm * 32) * 40, 40);
    v16h a1  = load_a_frag_lds(Ab + (wm * 32 + 16) * 40, 40);
    v16h b10 = load_b_frag_lds(B1b + (wn * 32) * 40, 40);
    v16h b11 = load_b_frag_lds(B1b + (wn * 32 + 16) * 40, 40);
    v16h b30 = load_b_frag_lds(B3b + (wn * 32) * 40, 40);
    v16h b31 = load_b_frag_lds(B3b + (wn * 32 + 16) * 40, 40);
    acc1[0][0] = wmma_f16(a0, b10, acc1[0][0]);
    acc1[0][1] = wmma_f16(a0, b11, acc1[0][1]);
    acc1[1][0] = wmma_f16(a1, b10, acc1[1][0]);
    acc1[1][1] = wmma_f16(a1, b11, acc1[1][1]);
    acc3[0][0] = wmma_f16(a0, b30, acc3[0][0]);
    acc3[0][1] = wmma_f16(a0, b31, acc3[0][1]);
    acc3[1][0] = wmma_f16(a1, b30, acc3[1][0]);
    acc3[1][1] = wmma_f16(a1, b31, acc3[1][1]);
    async_wait0();
    __syncthreads();
    p ^= 1;
  }
}

// ---- GEMM kernels ----------------------------------------------------------
__launch_bounds__(256)
__global__ void k_gemm_f16out(const _Float16* __restrict__ A, int lda,
                              const float* __restrict__ Bm, int ldb,
                              _Float16* __restrict__ C, int ldc, int K) {
  __shared__ __align__(16) _Float16 As[2 * ABUF_];
  __shared__ __align__(16) _Float16 Bs[2 * BBUF_];
  int rowBase = blockIdx.y * 128, colBase = blockIdx.x * 64;
  v8f acc[2][2] = {};
  gemm_core(A, lda, Bm, ldb, K, rowBase, colBase, As, Bs, acc);
  const int lane = threadIdx.x & 31, w = threadIdx.x >> 5;
  const int wm = w >> 1, wn = w & 1;
  const int rbase = (lane >> 4) << 3;
#pragma unroll
  for (int i = 0; i < 2; ++i)
#pragma unroll
    for (int j = 0; j < 2; ++j) {
      int row0 = rowBase + wm * 32 + i * 16 + rbase;
      int col  = colBase + wn * 32 + j * 16 + (lane & 15);
#pragma unroll
      for (int r = 0; r < 8; ++r)
        C[(size_t)(row0 + r) * ldc + col] = (_Float16)acc[i][j][r];
    }
}

__launch_bounds__(256)
__global__ void k_gemm_wo_res(const _Float16* __restrict__ A,
                              const float* __restrict__ Bm,
                              const float* __restrict__ resid,
                              float* __restrict__ Cout) {
  __shared__ __align__(16) _Float16 As[2 * ABUF_];
  __shared__ __align__(16) _Float16 Bs[2 * BBUF_];
  int rowBase = blockIdx.y * 128, colBase = blockIdx.x * 64;
  v8f acc[2][2] = {};
  gemm_core(A, NH_ * HD_, Bm, H_, NH_ * HD_, rowBase, colBase, As, Bs, acc);
  const int lane = threadIdx.x & 31, w = threadIdx.x >> 5;
  const int wm = w >> 1, wn = w & 1;
  const int rbase = (lane >> 4) << 3;
#pragma unroll
  for (int i = 0; i < 2; ++i)
#pragma unroll
    for (int j = 0; j < 2; ++j) {
      int row0 = rowBase + wm * 32 + i * 16 + rbase;
      int col  = colBase + wn * 32 + j * 16 + (lane & 15);
#pragma unroll
      for (int r = 0; r < 8; ++r) {
        size_t idx = (size_t)(row0 + r) * H_ + col;
        Cout[idx] = acc[i][j][r] + resid[idx];
      }
    }
}

__launch_bounds__(256)
__global__ void k_gemm_w13(const _Float16* __restrict__ A,
                           const float* __restrict__ W1m,
                           const float* __restrict__ W3m,
                           _Float16* __restrict__ act) {
  __shared__ __align__(16) _Float16 As[2 * ABUF_];
  __shared__ __align__(16) _Float16 Bs1[2 * BBUF_];
  __shared__ __align__(16) _Float16 Bs3[2 * BBUF_];
  int rowBase = blockIdx.y * 128, colBase = blockIdx.x * 64;
  v8f a1[2][2] = {}, a3[2][2] = {};
  gemm_core2(A, H_, W1m, W3m, F_, H_, rowBase, colBase, As, Bs1, Bs3, a1, a3);
  const int lane = threadIdx.x & 31, w = threadIdx.x >> 5;
  const int wm = w >> 1, wn = w & 1;
  const int rbase = (lane >> 4) << 3;
#pragma unroll
  for (int i = 0; i < 2; ++i)
#pragma unroll
    for (int j = 0; j < 2; ++j) {
      int row0 = rowBase + wm * 32 + i * 16 + rbase;
      int col  = colBase + wn * 32 + j * 16 + (lane & 15);
#pragma unroll
      for (int r = 0; r < 8; ++r) {
        float x = a1[i][j][r];
        float g = x / (1.f + __expf(-x));       // silu
        act[(size_t)(row0 + r) * F_ + col] = (_Float16)(g * a3[i][j][r]);
      }
    }
}

__launch_bounds__(256)
__global__ void k_gemm_w2acc(const _Float16* __restrict__ A,
                             const float* __restrict__ W2m,
                             const float* __restrict__ wfull, int e,
                             float* __restrict__ moe) {
  __shared__ __align__(16) _Float16 As[2 * ABUF_];
  __shared__ __align__(16) _Float16 Bs[2 * BBUF_];
  int rowBase = blockIdx.y * 128, colBase = blockIdx.x * 64;
  v8f acc[2][2] = {};
  gemm_core(A, F_, W2m, H_, F_, rowBase, colBase, As, Bs, acc);
  const int lane = threadIdx.x & 31, w = threadIdx.x >> 5;
  const int wm = w >> 1, wn = w & 1;
  const int rbase = (lane >> 4) << 3;
#pragma unroll
  for (int i = 0; i < 2; ++i)
#pragma unroll
    for (int j = 0; j < 2; ++j) {
      int row0 = rowBase + wm * 32 + i * 16 + rbase;
      int col  = colBase + wn * 32 + j * 16 + (lane & 15);
#pragma unroll
      for (int r = 0; r < 8; ++r) {
        int row = row0 + r;
        float rw = wfull[(size_t)row * E_ + e];
        moe[(size_t)row * H_ + col] += rw * acc[i][j][r];
      }
    }
}

// ---- RMSNorm (f32 in -> f16 out), one block per token ---------------------
__launch_bounds__(256)
__global__ void k_rmsnorm(const float* __restrict__ x,
                          const float* __restrict__ w,
                          _Float16* __restrict__ out) {
  const int t = blockIdx.x;
  const int tid = threadIdx.x;
  const float* xr = x + (size_t)t * H_;
  float v[8];
  float ss = 0.f;
#pragma unroll
  for (int j = 0; j < 8; ++j) {
    v[j] = xr[tid * 8 + j];
    ss += v[j] * v[j];
  }
#pragma unroll
  for (int m = 1; m < 32; m <<= 1) ss += __shfl_xor(ss, m, 32);
  __shared__ float red[8];
  __shared__ float s_r;
  if ((tid & 31) == 0) red[tid >> 5] = ss;
  __syncthreads();
  if (tid == 0) {
    float tot = 0.f;
    for (int i = 0; i < 8; ++i) tot += red[i];
    s_r = rsqrtf(tot / (float)H_ + EPS_);
  }
  __syncthreads();
  float rinv = s_r;
#pragma unroll
  for (int j = 0; j < 8; ++j) {
    int c = tid * 8 + j;
    out[(size_t)t * H_ + c] = (_Float16)(v[j] * rinv * w[c]);
  }
}

// ---- RoPE on q (16 heads) and k (4 heads), in place on f16 qkv ------------
__global__ void k_rope(_Float16* __restrict__ qkv, const int* __restrict__ positions) {
  int idx = blockIdx.x * blockDim.x + threadIdx.x;
  if (idx >= T_ * 20 * 64) return;
  int i  = idx & 63;
  int hh = (idx >> 6) % 20;
  int t  = idx / (20 * 64);
  int cb = (hh < NH_) ? hh * HD_ : NH_ * HD_ + (hh - NH_) * HD_;
  _Float16* p = qkv + (size_t)t * QKVN_ + cb;
  float pos = (float)positions[t];
  float inv = __powf(THETA_, -((float)i) / 64.f);
  float ang = pos * inv;
  float c = __cosf(ang), s = __sinf(ang);
  float x1 = (float)p[i], x2 = (float)p[i + 64];
  p[i]      = (_Float16)(x1 * c - x2 * s);
  p[i + 64] = (_Float16)(x1 * s + x2 * c);
}

// ---- Flash attention: one wave per (b, h, 16-query tile), causal ----------
__launch_bounds__(32)
__global__ void k_attn(const _Float16* __restrict__ qkv,
                       _Float16* __restrict__ out) {
  __shared__ __align__(8) _Float16 Pbuf[16 * 36];
  const int lane = threadIdx.x;
  const int q0 = blockIdx.x * 16;
  const int h  = blockIdx.y;
  const int b  = blockIdx.z;
  const int kvh = h >> 2;                 // GQA: rep = NH/NKV = 4
  const int m  = lane & 15;
  const int kb = (lane >> 4) * 8;
  const int rbase = (lane >> 4) << 3;
  const float scale = 0.08838834764831845f; // 1/sqrt(128), folded into Q

  // Q fragments, 4 chunks of 32 along head-dim, pre-scaled
  FragA qf[4];
  {
    const _Float16* qrow = qkv + (size_t)(b * S_ + q0 + m) * QKVN_ + h * HD_;
#pragma unroll
    for (int c = 0; c < 4; ++c)
#pragma unroll
      for (int j = 0; j < 8; ++j) {
        int k = c * 32 + ((j >> 2) << 4) + kb + ((j & 3) << 1);
        qf[c].h[2 * j]     = (_Float16)((float)qrow[k] * scale);
        qf[c].h[2 * j + 1] = (_Float16)((float)qrow[k + 1] * scale);
      }
  }

  v8f zero8 = {};
  v8f o[8];
#pragma unroll
  for (int nt = 0; nt < 8; ++nt) o[nt] = zero8;
  float mrow[8], lrow[8];
#pragma unroll
  for (int r = 0; r < 8; ++r) { mrow[r] = -1e30f; lrow[r] = 0.f; }

  for (int kc = 0; kc < q0 + 16; kc += 32) {
    // ---- scores: two 16-key D tiles via 4 WMMAs each ----
    v8f s[2];
    s[0] = zero8; s[1] = zero8;
#pragma unroll
    for (int half = 0; half < 2; ++half) {
      const int key = kc + half * 16 + (lane & 15);
      const _Float16* kp = qkv + (size_t)(b * S_ + key) * QKVN_
                         + NH_ * HD_ + kvh * HD_ + ((lane >> 4) * 8);
#pragma unroll
      for (int c = 0; c < 4; ++c) {
        FragA bf;
#pragma unroll
        for (int j = 0; j < 8; ++j)
          bf.u[j] = *reinterpret_cast<const uint32_t*>(kp + c * 32 + ((j >> 2) << 4) + ((j & 3) << 1));
        s[half] = wmma_f16(qf[c].v, bf.v, s[half]);
      }
    }
    // ---- causal mask + online softmax + stage P into LDS ----
#pragma unroll
    for (int r = 0; r < 8; ++r) {
      int qi   = q0 + rbase + r;
      int key0 = kc + (lane & 15);
      float s0 = (key0      <= qi) ? s[0][r] : -1e9f;
      float s1 = (key0 + 16 <= qi) ? s[1][r] : -1e9f;
      float mv = fmaxf(s0, s1);
#pragma unroll
      for (int msk = 1; msk < 16; msk <<= 1) mv = fmaxf(mv, __shfl_xor(mv, msk, 32));
      float mn = fmaxf(mrow[r], mv);
      float p0 = __expf(s0 - mn), p1 = __expf(s1 - mn);
      float rs = p0 + p1;
#pragma unroll
      for (int msk = 1; msk < 16; msk <<= 1) rs += __shfl_xor(rs, msk, 32);
      float alpha = __expf(mrow[r] - mn);
      mrow[r] = mn;
      lrow[r] = lrow[r] * alpha + rs;
#pragma unroll
      for (int nt = 0; nt < 8; ++nt) o[nt][r] *= alpha;
      _Float16* pr = Pbuf + (rbase + r) * 36 + (lane & 15);
      pr[0]  = (_Float16)p0;
      pr[16] = (_Float16)p1;
    }
    wave_lds_fence();
    // ---- P as A fragment (16x32) ----
    FragA pa;
    {
      const _Float16* rp = Pbuf + m * 36;
#pragma unroll
      for (int j = 0; j < 8; ++j) {
        int k = ((j >> 2) << 4) + kb + ((j & 3) << 1);
        pa.u[j] = *reinterpret_cast<const uint32_t*>(rp + k);
      }
    }
    // ---- PV: 8 WMMAs across head-dim ----
#pragma unroll
    for (int nt = 0; nt < 8; ++nt) {
      FragA vf;
      const int hd0 = nt * 16 + (lane & 15);
      const size_t vbase = (size_t)(b * S_) * QKVN_ + (NH_ + NKV_) * HD_
                         + kvh * HD_ + hd0;
#pragma unroll
      for (int j = 0; j < 8; ++j) {
        int key = kc + ((lane >> 4) * 8) + ((j >> 2) << 4) + ((j & 3) << 1);
        vf.h[2 * j]     = qkv[vbase + (size_t)key * QKVN_];
        vf.h[2 * j + 1] = qkv[vbase + (size_t)(key + 1) * QKVN_];
      }
      o[nt] = wmma_f16(pa.v, vf.v, o[nt]);
    }
    wave_lds_fence();
  }

#pragma unroll
  for (int nt = 0; nt < 8; ++nt) {
    int col = h * HD_ + nt * 16 + (lane & 15);
#pragma unroll
    for (int r = 0; r < 8; ++r) {
      int row = b * S_ + q0 + rbase + r;
      out[(size_t)row * (NH_ * HD_) + col] = (_Float16)(o[nt][r] / lrow[r]);
    }
  }
}

// ---- gate: softmax over E=8, top-2, normalized routing weights ------------
__launch_bounds__(32)
__global__ void k_gate(const _Float16* __restrict__ x,
                       const float* __restrict__ gw,
                       float* __restrict__ wfull) {
  int t = blockIdx.x;
  int lane = threadIdx.x;
  float acc[E_];
#pragma unroll
  for (int e = 0; e < E_; ++e) acc[e] = 0.f;
  const _Float16* xr = x + (size_t)t * H_;
  for (int hh = lane; hh < H_; hh += 32) {
    float xv = (float)xr[hh];
    const float* g = gw + (size_t)hh * E_;
#pragma unroll
    for (int e = 0; e < E_; ++e) acc[e] += xv * g[e];
  }
#pragma unroll
  for (int e = 0; e < E_; ++e)
#pragma unroll
    for (int m = 1; m < 32; m <<= 1) acc[e] += __shfl_xor(acc[e], m, 32);
  if (lane == 0) {
    float mx = acc[0];
    for (int e = 1; e < E_; ++e) mx = fmaxf(mx, acc[e]);
    float p[E_], sum = 0.f;
    for (int e = 0; e < E_; ++e) { p[e] = __expf(acc[e] - mx); sum += p[e]; }
    for (int e = 0; e < E_; ++e) p[e] /= sum;
    int i0 = 0;
    for (int e = 1; e < E_; ++e) if (p[e] > p[i0]) i0 = e;
    int i1 = (i0 == 0) ? 1 : 0;
    for (int e = 0; e < E_; ++e) if (e != i0 && p[e] > p[i1]) i1 = e;
    float norm = p[i0] + p[i1];
    float* wr = wfull + (size_t)t * E_;
    for (int e = 0; e < E_; ++e) wr[e] = 0.f;
    wr[i0] = p[i0] / norm;
    wr[i1] = p[i1] / norm;
  }
}

__global__ void k_zero(float* __restrict__ p, int n) {
  int i = blockIdx.x * blockDim.x + threadIdx.x;
  if (i < n) p[i] = 0.f;
}

// ---- host orchestration ----------------------------------------------------
extern "C" void kernel_launch(void* const* d_in, const int* in_sizes, int n_in,
                              void* d_out, int out_size, void* d_ws, size_t ws_size,
                              hipStream_t stream) {
  (void)in_sizes; (void)n_in; (void)out_size; (void)ws_size;
  const int*   positions = (const int*)d_in[0];
  const float* hidden    = (const float*)d_in[1];
  const float* ln1_w     = (const float*)d_in[2];
  const float* ln2_w     = (const float*)d_in[3];
  const float* wqkv      = (const float*)d_in[4];
  const float* wo        = (const float*)d_in[5];
  const float* gate_w    = (const float*)d_in[6];
  const float* w1        = (const float*)d_in[7];
  const float* w2        = (const float*)d_in[8];
  const float* w3        = (const float*)d_in[9];

  float* moe_out = (float*)d_out;
  float* resid2  = (float*)d_out + (size_t)T_ * H_;

  char* ws = (char*)d_ws;
  size_t off = 0;
  auto alloc = [&](size_t bytes) {
    char* p = ws + off;
    off = (off + bytes + 255) & ~(size_t)255;
    return p;
  };
  _Float16* x1h  = (_Float16*)alloc((size_t)T_ * H_ * 2);      // normed input
  _Float16* qkvh = (_Float16*)alloc((size_t)T_ * QKVN_ * 2);   // qkv (f16)
  _Float16* aoh  = (_Float16*)alloc((size_t)T_ * NH_ * HD_ * 2);
  _Float16* x2h  = (_Float16*)alloc((size_t)T_ * H_ * 2);
  _Float16* acth = (_Float16*)alloc((size_t)T_ * F_ * 2);      // per-expert act
  float*    wful = (float*)alloc((size_t)T_ * E_ * 4);

  // 1. RMSNorm 1
  k_rmsnorm<<<T_, 256, 0, stream>>>(hidden, ln1_w, x1h);
  // 2. QKV GEMM  [T,H] x [H,3072]
  k_gemm_f16out<<<dim3(QKVN_ / 64, T_ / 128), 256, 0, stream>>>(
      x1h, H_, wqkv, QKVN_, qkvh, QKVN_, H_);
  // 3. RoPE on q + k
  {
    int n = T_ * 20 * 64;
    k_rope<<<(n + 255) / 256, 256, 0, stream>>>(qkvh, positions);
  }
  // 4. causal flash attention
  k_attn<<<dim3(S_ / 16, NH_, B_), 32, 0, stream>>>(qkvh, aoh);
  // 5. WO GEMM + residual -> residual2 (second output)
  k_gemm_wo_res<<<dim3(H_ / 64, T_ / 128), 256, 0, stream>>>(aoh, wo, hidden, resid2);
  // 6. RMSNorm 2
  k_rmsnorm<<<T_, 256, 0, stream>>>(resid2, ln2_w, x2h);
  // 7. gate + top-2 routing
  k_gate<<<T_, 32, 0, stream>>>(x2h, gate_w, wful);
  // 8. MoE (dense over experts, routed accumulate)
  {
    int n = T_ * H_;
    k_zero<<<(n + 255) / 256, 256, 0, stream>>>(moe_out, n);
  }
  for (int e = 0; e < E_; ++e) {
    k_gemm_w13<<<dim3(F_ / 64, T_ / 128), 256, 0, stream>>>(
        x2h, w1 + (size_t)e * H_ * F_, w3 + (size_t)e * H_ * F_, acth);
    k_gemm_w2acc<<<dim3(H_ / 64, T_ / 128), 256, 0, stream>>>(
        acth, w2 + (size_t)e * F_ * H_, wful, e, moe_out);
  }
}
